// LinearAttention1D_137438953829
// MI455X (gfx1250) — matrix operand order repaired and checked by reading, not compiled, not run on hardware
//
#include <hip/hip_runtime.h>
#include <hip/hip_fp16.h>

#define NB 4
#define NCH 128
#define NSEQ 65536
#define NH 4
#define DHEAD 32
#define NQKV 384
#define SCALE_Q 0.42044820762685725f   // 1/sqrt(sqrt(32))
#define GN_EPS 1e-5f

typedef _Float16 f16;
typedef _Float16 f16x8  __attribute__((ext_vector_type(8)));
typedef _Float16 f16x16 __attribute__((ext_vector_type(16)));
typedef float    f32x8  __attribute__((ext_vector_type(8)));
typedef float    f32x4  __attribute__((ext_vector_type(4)));
typedef unsigned int u32x4 __attribute__((ext_vector_type(4)));
typedef int      i32x4  __attribute__((ext_vector_type(4)));
typedef int      i32x8  __attribute__((ext_vector_type(8)));

static __device__ __forceinline__ f32x8 wmma_f16(f16x16 a, f16x16 b, f32x8 c) {
  // (neg_a, A, neg_b, B, c_mod, C, reuse_a, reuse_b) -> v_wmma_f32_16x16x32_f16
  return __builtin_amdgcn_wmma_f32_16x16x32_f16(false, a, false, b, (short)0, c, false, false);
}

static __device__ __forceinline__ f16x16 cat16(f16x8 lo, f16x8 hi) {
  f16x16 r;
#pragma unroll
  for (int i = 0; i < 8; ++i) { r[i] = lo[i]; r[i + 8] = hi[i]; }
  return r;
}

// LDS pointers are 32-bit workgroup-relative offsets in addrspace(3); the TDM
// descriptor's lds_addr field wants exactly that byte offset.
static __device__ __forceinline__ unsigned lds_off(const void* p) {
  return (unsigned)(unsigned long long)
      (const __attribute__((address_space(3))) char*)(const char*)p;
}

// order-preserving float->uint encoding so uint atomicMax == float max;
// key 0 is below every finite key -> memset(0) initializes to "-inf".
static __device__ __forceinline__ unsigned fenc(float x) {
  unsigned u = __float_as_uint(x);
  return (u & 0x80000000u) ? ~u : (u | 0x80000000u);
}
static __device__ __forceinline__ float fdec(unsigned k) {
  unsigned u = (k & 0x80000000u) ? (k & 0x7fffffffu) : ~k;
  return __uint_as_float(u);
}

// Issue a 2D f16 tile load (rows x cols, row stride in halfs) into LDS via the
// Tensor Data Mover, then wait on TENSORcnt. Wave-level op: call from ONE wave.
static __device__ __forceinline__ void tdm_load_f16_2d(
    const f16* gsrc, unsigned ldsByteOff, int rows, int cols, int rowStride) {
  unsigned long long ga = (unsigned long long)gsrc;
  u32x4 g0;
  g0[0] = 1u;                                        // count=1 (valid user D#)
  g0[1] = ldsByteOff;                                // lds_addr
  g0[2] = (unsigned)(ga & 0xffffffffu);              // global_addr[31:0]
  g0[3] = (unsigned)((ga >> 32) & 0x01ffffffu) | (2u << 30);  // addr[56:32], type=2
  i32x8 g1;
  g1[0] = (1 << 16);                                 // wg_mask=0, data_size=1 (2B)
  g1[1] = ((rowStride & 0xffff) << 16);              // tensor_dim0[15:0] @ bits63:48
  g1[2] = ((unsigned)rowStride >> 16) | (rows << 16);// tensor_dim0[31:16], tensor_dim1[15:0]
  g1[3] = (cols << 16);                              // tensor_dim1[31:16]=0, tile_dim0
  g1[4] = rows;                                      // tile_dim1, tile_dim2=0
  g1[5] = rowStride;                                 // tensor_dim0_stride[31:0]
  g1[6] = 0;                                         // stride0[47:32], stride1[15:0]
  g1[7] = 0;                                         // tensor_dim1_stride[47:16]
  i32x4 g2 = {0, 0, 0, 0};
  i32x4 g3 = {0, 0, 0, 0};
#if defined(__clang_major__) && (__clang_major__ >= 23)
  i32x8 g4 = {0, 0, 0, 0, 0, 0, 0, 0};
  __builtin_amdgcn_tensor_load_to_lds(g0, g1, g2, g3, g4, 0);
#else
  __builtin_amdgcn_tensor_load_to_lds(g0, g1, g2, g3, 0);
#endif
  __builtin_amdgcn_s_wait_tensorcnt(0);
}

// ---------------------------------------------------------------------------
// K1: qkv = W_qkv @ x (f16 WMMA, fp32 acc); fused q-softmax(over d) * SCALE;
//     k,v stored f16; per-channel running max of k via uint atomicMax.
// block = 256 threads (8 waves), handles (b, 128 columns). LDS pitch 136.
// ---------------------------------------------------------------------------
__global__ __launch_bounds__(256) void k1_qkv(
    const float* __restrict__ x, const float* __restrict__ w_qkv,
    f16* __restrict__ qws, f16* __restrict__ kws, f16* __restrict__ vws,
    unsigned* __restrict__ kmaxu) {
  extern __shared__ char smem[];
  f16* wlds = (f16*)smem;                         // [384][136] f16
  f16* xT   = (f16*)(smem + NQKV * 136 * 2);      // [128 cols][136] f16 (transposed)
  const int tid = threadIdx.x;
  const int b   = blockIdx.y;
  const int n0  = blockIdx.x * 128;

  for (int i = tid; i < NQKV * NCH; i += 256)
    wlds[(i >> 7) * 136 + (i & 127)] = (f16)w_qkv[i];

  for (int i = tid; i < (NCH * 128) / 4; i += 256) {
    int c = i >> 5;              // 32 float4 chunks per channel row
    int j = (i & 31) * 4;
    f32x4 v = *(const f32x4*)(x + (((size_t)(b * NCH + c)) << 16) + n0 + j);
#pragma unroll
    for (int t = 0; t < 4; ++t) xT[(j + t) * 136 + c] = (f16)v[t];
  }
  __syncthreads();

  const int wave = tid >> 5, lane = tid & 31;
  const int hi = lane >> 4, lm = lane & 15;
  const int coln = n0 + 16 * wave + lm;           // this lane's output column

  auto loadA = [&](int R, int kk) -> f16x16 {     // A 16x32 f16 fragment
    const f16* p = &wlds[(R + lm) * 136 + kk * 32 + (hi ? 8 : 0)];
    return cat16(*(const f16x8*)p, *(const f16x8*)(p + 16));
  };
  auto loadB = [&](int kk) -> f16x16 {            // B 32x16 f16 fragment
    const f16* p = &xT[(16 * wave + lm) * 136 + kk * 32 + (hi ? 8 : 0)];
    return cat16(*(const f16x8*)p, *(const f16x8*)(p + 16));
  };

  // ---- q rows 0..127: per-column softmax over each head's 32 channels ----
  for (int h = 0; h < NH; ++h) {
    f32x8 a0 = {}, a1 = {};
#pragma unroll
    for (int kk = 0; kk < 4; ++kk) {
      f16x16 bb = loadB(kk);
      a0 = wmma_f16(loadA(32 * h, kk), bb, a0);
      a1 = wmma_f16(loadA(32 * h + 16, kk), bb, a1);
    }
    float m = -3.4e38f;
#pragma unroll
    for (int j = 0; j < 8; ++j) m = fmaxf(m, fmaxf(a0[j], a1[j]));
    m = fmaxf(m, __shfl_xor(m, 16));              // lanes L,L^16 cover all 32 d
    float e0[8], e1[8], s = 0.f;
#pragma unroll
    for (int j = 0; j < 8; ++j) {
      e0[j] = __expf(a0[j] - m); e1[j] = __expf(a1[j] - m);
      s += e0[j] + e1[j];
    }
    s += __shfl_xor(s, 16);
    float inv = SCALE_Q / s;
#pragma unroll
    for (int j = 0; j < 8; ++j) {
      int r0 = 32 * h + j + 8 * hi;
      qws[(((size_t)(b * NCH + r0)) << 16) + coln]      = (f16)(e0[j] * inv);
      qws[(((size_t)(b * NCH + r0 + 16)) << 16) + coln] = (f16)(e1[j] * inv);
    }
  }

  // ---- k rows 128..255: store f16 + per-channel max ----
  for (int rt = 0; rt < 8; ++rt) {
    f32x8 acc = {};
#pragma unroll
    for (int kk = 0; kk < 4; ++kk)
      acc = wmma_f16(loadA(128 + 16 * rt, kk), loadB(kk), acc);
#pragma unroll
    for (int j = 0; j < 8; ++j) {
      int kr = 16 * rt + j + 8 * hi;
      kws[(((size_t)(b * NCH + kr)) << 16) + coln] = (f16)acc[j];
    }
#pragma unroll
    for (int j = 0; j < 8; ++j) {                 // row-max over 16 columns
      float mv = acc[j];
      mv = fmaxf(mv, __shfl_xor(mv, 1));
      mv = fmaxf(mv, __shfl_xor(mv, 2));
      mv = fmaxf(mv, __shfl_xor(mv, 4));
      mv = fmaxf(mv, __shfl_xor(mv, 8));
      if (lm == 0)
        atomicMax(&kmaxu[b * NCH + 16 * rt + j + 8 * hi], fenc(mv));
    }
  }

  // ---- v rows 256..383: store f16 ----
  for (int rt = 0; rt < 8; ++rt) {
    f32x8 acc = {};
#pragma unroll
    for (int kk = 0; kk < 4; ++kk)
      acc = wmma_f16(loadA(256 + 16 * rt, kk), loadB(kk), acc);
#pragma unroll
    for (int j = 0; j < 8; ++j) {
      int vr = 16 * rt + j + 8 * hi;
      vws[(((size_t)(b * NCH + vr)) << 16) + coln] = (f16)acc[j];
    }
  }
}

// ---------------------------------------------------------------------------
// K2: per (b,h,split): ctx[d,e] += sum_n exp(k[d,n]-max[d]) * v[e,n] (WMMA,
// K = n in chunks of 32) and Z[d] += sum_n exp(...). 1024 waves total.
// ---------------------------------------------------------------------------
__global__ __launch_bounds__(256) void k2_context(
    const f16* __restrict__ kws, const f16* __restrict__ vws,
    const unsigned* __restrict__ kmaxu,
    float* __restrict__ ctx, float* __restrict__ Z) {
  const int tid = threadIdx.x;
  const int gw = blockIdx.x * 8 + (tid >> 5);     // 0..1023
  const int bh = gw >> 6, split = gw & 63;        // 16 (b,h) x 64 splits
  const int b = bh >> 2, h = bh & 3;
  const int lane = tid & 31, hi = lane >> 4, lm = lane & 15;

  const f16* kb = kws + (((size_t)(b * NCH + h * DHEAD)) << 16);
  const f16* vb = vws + (((size_t)(b * NCH + h * DHEAD)) << 16);
  const float km0 = fdec(kmaxu[b * NCH + h * DHEAD + lm]);
  const float km1 = fdec(kmaxu[b * NCH + h * DHEAD + 16 + lm]);

  f32x8 c00 = {}, c01 = {}, c10 = {}, c11 = {};
  float z0 = 0.f, z1 = 0.f;

  for (int cc = 0; cc < 32; ++cc) {
    const int n0 = split * 1024 + cc * 32;
    if (cc + 4 < 32) {                            // global_prefetch_b8
      __builtin_prefetch(&kb[((size_t)lm << 16) + n0 + 128], 0, 0);
      __builtin_prefetch(&vb[((size_t)lm << 16) + n0 + 128], 0, 0);
    }
    const int nb = n0 + (hi ? 8 : 0);
    f16x8 r0 = *(const f16x8*)&kb[((size_t)lm << 16) + nb];
    f16x8 r1 = *(const f16x8*)&kb[((size_t)lm << 16) + nb + 16];
    f16x8 s0 = *(const f16x8*)&kb[((size_t)(16 + lm) << 16) + nb];
    f16x8 s1 = *(const f16x8*)&kb[((size_t)(16 + lm) << 16) + nb + 16];
    f16x16 a0, a1;
#pragma unroll
    for (int t = 0; t < 8; ++t) {
      float e00 = __expf((float)r0[t] - km0);
      float e01 = __expf((float)r1[t] - km0);
      float e10 = __expf((float)s0[t] - km1);
      float e11 = __expf((float)s1[t] - km1);
      a0[t] = (f16)e00; a0[t + 8] = (f16)e01;
      a1[t] = (f16)e10; a1[t + 8] = (f16)e11;
      z0 += e00 + e01;  z1 += e10 + e11;
    }
    const int nb2 = n0 + (hi ? 8 : 0);            // B = V^T, table order in n
    const f16* p0 = &vb[((size_t)lm << 16) + nb2];
    const f16* p1 = &vb[((size_t)(16 + lm) << 16) + nb2];
    f16x16 b0 = cat16(*(const f16x8*)p0, *(const f16x8*)(p0 + 16));
    f16x16 b1 = cat16(*(const f16x8*)p1, *(const f16x8*)(p1 + 16));
    c00 = wmma_f16(a0, b0, c00);
    c01 = wmma_f16(a0, b1, c01);
    c10 = wmma_f16(a1, b0, c10);
    c11 = wmma_f16(a1, b1, c11);
  }

  z0 += __shfl_xor(z0, 16);
  z1 += __shfl_xor(z1, 16);
  if (hi == 0) {
    atomicAdd(&Z[b * NCH + h * DHEAD + lm], z0);
    atomicAdd(&Z[b * NCH + h * DHEAD + 16 + lm], z1);
  }
  float* cbase = ctx + (size_t)bh * DHEAD * DHEAD;
#pragma unroll
  for (int j = 0; j < 8; ++j) {
    int d0 = j + 8 * hi;
    atomicAdd(&cbase[d0 * DHEAD + lm],             c00[j]);
    atomicAdd(&cbase[d0 * DHEAD + 16 + lm],        c01[j]);
    atomicAdd(&cbase[(16 + d0) * DHEAD + lm],      c10[j]);
    atomicAdd(&cbase[(16 + d0) * DHEAD + 16 + lm], c11[j]);
  }
}

// ---------------------------------------------------------------------------
// K2b: W2[b][o][h*32+d] = sum_e w_out[o,h*32+e] * ctx[b,h,d,e] / Z[b,h*32+d]
// ---------------------------------------------------------------------------
__global__ __launch_bounds__(256) void k2b_w2(
    const float* __restrict__ w_out, const float* __restrict__ ctx,
    const float* __restrict__ Z, f16* __restrict__ W2) {
  const int b = blockIdx.x;
  for (int i = threadIdx.x; i < NCH * NCH; i += 256) {
    int o = i >> 7, col = i & 127;
    int h = col >> 5, d = col & 31;
    const float* cp = ctx + ((size_t)(b * NH + h) * DHEAD + d) * DHEAD;
    const float* wp = w_out + o * NCH + h * DHEAD;
    float s = 0.f;
#pragma unroll
    for (int e = 0; e < DHEAD; ++e) s += wp[e] * cp[e];
    W2[(size_t)b * NCH * NCH + i] = (f16)(s / Z[b * NCH + col]);
  }
}

// ---------------------------------------------------------------------------
// K3: y_pre = W2[b] @ q_sm + b_out (WMMA); q tile staged by the Tensor Data
// Mover (TENSOR_LOAD_TO_LDS + s_wait_tensorcnt), then transposed in LDS.
// Fused GroupNorm-stat accumulation.
// ---------------------------------------------------------------------------
__global__ __launch_bounds__(256) void k3_out(
    const f16* __restrict__ W2, const f16* __restrict__ qws,
    const float* __restrict__ b_out, float* __restrict__ ypre,
    float* __restrict__ stats) {
  extern __shared__ char smem[];
  f16* wl   = (f16*)smem;                         // [128][136] repacked W2
  f16* qT   = (f16*)(smem + NCH * 136 * 2);       // [128 cols][136] transposed q
  f16* qrow = (f16*)(smem + 2 * NCH * 136 * 2);   // [128][128] TDM landing buffer
  __shared__ float red[16];
  const int tid = threadIdx.x;
  const int b   = blockIdx.y;
  const int n0  = blockIdx.x * 128;

  // Wave 0 issues one TDM 2D-tile load: q[b, 0..127, n0..n0+127] -> qrow.
  if (tid < 32) {
    tdm_load_f16_2d(qws + (((size_t)(b * NCH)) << 16) + n0,
                    lds_off(qrow), NCH, 128, NSEQ);
  }

  for (int i = tid; i < NCH * NCH; i += 256)
    wl[(i >> 7) * 136 + (i & 127)] = W2[(size_t)b * NCH * NCH + i];
  __syncthreads();                                // TDM data visible to all waves

  for (int i = tid; i < (NCH * 128) / 8; i += 256) {
    int c = i >> 4, j = (i & 15) * 8;
    f16x8 v = *(const f16x8*)(qrow + c * 128 + j);
#pragma unroll
    for (int t = 0; t < 8; ++t) qT[(j + t) * 136 + c] = v[t];
  }
  __syncthreads();

  const int wave = tid >> 5, lane = tid & 31;
  const int hi = lane >> 4, lm = lane & 15;
  const int coln = n0 + 16 * wave + lm;

  float ls = 0.f, lq = 0.f;
  for (int rt = 0; rt < 8; ++rt) {
    f32x8 acc = {};
#pragma unroll
    for (int kk = 0; kk < 4; ++kk) {
      const f16* ap = &wl[(16 * rt + lm) * 136 + kk * 32 + (hi ? 8 : 0)];
      f16x16 a = cat16(*(const f16x8*)ap, *(const f16x8*)(ap + 16));
      const f16* bp = &qT[(16 * wave + lm) * 136 + kk * 32 + (hi ? 8 : 0)];
      f16x16 bb = cat16(*(const f16x8*)bp, *(const f16x8*)(bp + 16));
      acc = wmma_f16(a, bb, acc);
    }
#pragma unroll
    for (int j = 0; j < 8; ++j) {
      int o = 16 * rt + j + 8 * hi;
      float val = acc[j] + b_out[o];
      ypre[(((size_t)(b * NCH + o)) << 16) + coln] = val;
      ls += val; lq += val * val;
    }
  }
#pragma unroll
  for (int m = 1; m < 32; m <<= 1) { ls += __shfl_xor(ls, m); lq += __shfl_xor(lq, m); }
  if (lane == 0) { red[wave] = ls; red[8 + wave] = lq; }
  __syncthreads();
  if (tid == 0) {
    float S = 0.f, Q = 0.f;
#pragma unroll
    for (int w = 0; w < 8; ++w) { S += red[w]; Q += red[8 + w]; }
    atomicAdd(&stats[2 * b], S);
    atomicAdd(&stats[2 * b + 1], Q);
  }
}

// ---------------------------------------------------------------------------
// K4: GroupNorm apply (per-b mean/var from stats) + affine, vectorized x8.
// ---------------------------------------------------------------------------
__global__ __launch_bounds__(256) void k4_gn(
    const float* __restrict__ ypre, const float* __restrict__ stats,
    const float* __restrict__ gamma, const float* __restrict__ beta,
    float* __restrict__ out) {
  const size_t base = ((size_t)blockIdx.x * 256 + threadIdx.x) * 8;
  const int b = (int)(base >> 23);
  const int c = (int)(base >> 16) & 127;
  const float cnt = (float)NCH * (float)NSEQ;
  float mean = stats[2 * b] / cnt;
  float var  = stats[2 * b + 1] / cnt - mean * mean;
  float inv  = rsqrtf(var + GN_EPS);
  float g  = gamma[c] * inv;
  float be = beta[c] - mean * g;
  f32x4 v0 = *(const f32x4*)(ypre + base);
  f32x4 v1 = *(const f32x4*)(ypre + base + 4);
#pragma unroll
  for (int t = 0; t < 4; ++t) { v0[t] = v0[t] * g + be; v1[t] = v1[t] * g + be; }
  *(f32x4*)(out + base)     = v0;
  *(f32x4*)(out + base + 4) = v1;
}

// ---------------------------------------------------------------------------
extern "C" void kernel_launch(void* const* d_in, const int* in_sizes, int n_in,
                              void* d_out, int out_size, void* d_ws, size_t ws_size,
                              hipStream_t stream) {
  const float* x      = (const float*)d_in[0];
  const float* w_qkv  = (const float*)d_in[1];
  const float* w_out  = (const float*)d_in[2];
  const float* b_out  = (const float*)d_in[3];
  const float* gamma  = (const float*)d_in[4];
  const float* beta   = (const float*)d_in[5];

  const size_t PLANE = (size_t)NB * NCH * NSEQ;   // 33,554,432 elements
  f16*   qws   = (f16*)d_ws;
  f16*   kws   = qws + PLANE;
  f16*   vws   = kws + PLANE;
  float* ypre  = (float*)(vws + PLANE);
  float* ctx   = ypre + PLANE;                    // [B][H][32][32]
  float* Z     = ctx + NB * NH * DHEAD * DHEAD;   // [B][128]
  unsigned* kmaxu = (unsigned*)(Z + NB * NCH);    // [B][128]
  float* stats = (float*)(kmaxu + NB * NCH);      // [B][2]
  f16*   W2    = (f16*)(stats + 2 * NB);          // [B][128][128]

  // zero the accumulators (ctx + Z + kmax + stats are contiguous)
  hipMemsetAsync(ctx, 0,
                 (size_t)(NB * NH * DHEAD * DHEAD + NB * NCH + NB * NCH + 2 * NB) * 4,
                 stream);

  const size_t s1 = (size_t)(NQKV + NCH) * 136 * 2;   // 139,264 B dyn LDS
  k1_qkv<<<dim3(NSEQ / 128, NB), 256, s1, stream>>>(x, w_qkv, qws, kws, vws, kmaxu);
  k2_context<<<128, 256, 0, stream>>>(kws, vws, kmaxu, ctx, Z);
  k2b_w2<<<NB, 256, 0, stream>>>(w_out, ctx, Z, W2);
  const size_t s3 = (size_t)(2 * NCH * 136 + NCH * 128) * 2;  // 102,400 B dyn LDS
  k3_out<<<dim3(NSEQ / 128, NB), 256, s3, stream>>>(W2, qws, b_out, ypre, stats);
  k4_gn<<<(int)(PLANE / (256 * 8)), 256, 0, stream>>>(ypre, stats, gamma, beta,
                                                      (float*)d_out);
}
